// PointTransformerSegmentation_34239479284297
// MI455X (gfx1250) — compile-verified, matrix operand order checked by reading, not run on hardware
//
#include <hip/hip_runtime.h>
#include <hip/hip_bf16.h>

// ---------------------------------------------------------------------------
// PointTransformer segmentation forward for MI455X (gfx1250, wave32).
// All dense linears run through v_wmma_f32_16x16x32_f16 (f16 in, f32 accum).
// GEMM tiles are staged with CDNA5 async global->LDS copies (ASYNCcnt) and
// double-buffered LDS so WMMA overlaps the next tile's fill.
// Irregular ops (knn/fps/gather/softmax-agg/interp) are VALU kernels.
// ---------------------------------------------------------------------------

typedef __attribute__((ext_vector_type(16))) _Float16 v16h;
typedef __attribute__((ext_vector_type(8)))  _Float16 v8h;
typedef __attribute__((ext_vector_type(8)))  float    v8f;

static inline int kpad32(int k) { return (k + 31) & ~31; }

#define LDS_STRIDE 40   // halfs per LDS tile row (80B): 16B-aligned chunks, skews banks

// one 16-byte async copy: global (64-bit vaddr) -> LDS (32-bit byte address)
__device__ __forceinline__ void async_cp16(unsigned lds_addr, const void* gptr)
{
    asm volatile("global_load_async_to_lds_b128 %0, %1, off"
                 :: "v"(lds_addr), "v"((unsigned long long)(size_t)gptr)
                 : "memory");
}

// ---------------------------------------------------------------------------
// WMMA GEMM: out[M,N] = act(A[M,K] @ W[N,K]^T + bias)
//   A: f16 row-major, stride lda (== K, multiple of 32)
//   W: f16 row-major [N, K] (torch Linear layout), stride ldw (== K)
//   o32: optional f32 out (stride ldo32); o16: optional f16 out (stride ldo16)
// Block: 128 threads (4 waves). Tile: 64(M) x 64(N), K stepped by 32.
// Double-buffered LDS; each stage = 4 async b128 copies per thread (2 A, 2 W).
// Each wave owns a 16x64 strip: 4 accumulators; all 5 fragments are loaded
// before the MMA chain so the 4 WMMAs issue back-to-back (one dscnt wait).
// ---------------------------------------------------------------------------
__global__ __launch_bounds__(128)
void wmma_gemm_kernel(const _Float16* __restrict__ A, int lda,
                      const _Float16* __restrict__ W, int ldw,
                      const float* __restrict__ bias,
                      float* __restrict__ o32, int ldo32,
                      _Float16* __restrict__ o16, int ldo16,
                      int M, int N, int K, int relu)
{
    __shared__ __align__(16) _Float16 As[2][64 * LDS_STRIDE];
    __shared__ __align__(16) _Float16 Bs[2][64 * LDS_STRIDE];
    __shared__ __align__(16) char     Dump[128 * 32];   // sink for masked-off async lanes

    const int tid  = threadIdx.x;
    const int wave = tid >> 5;
    const int lane = tid & 31;
    const int half = lane >> 4;          // lane half selects K sub-range
    const int lr   = lane & 15;
    const int m0   = blockIdx.y * 64;
    const int n0   = blockIdx.x * 64;

    const int ldr = tid >> 1;            // 0..63: tile row this thread stages
    const int seg = (tid & 1) * 16;      // 0 or 16: K segment (16 halfs = 32B)

    const bool a_ok = (m0 + ldr) < M;
    const bool w_ok = (n0 + ldr) < N;

    // clamped global row pointers (row 0 for out-of-range: safe, data discarded)
    const _Float16* aptr = A + (size_t)(a_ok ? (m0 + ldr) : 0) * lda + seg;
    const _Float16* wptr = W + (size_t)(w_ok ? (n0 + ldr) : 0) * ldw + seg;

    // LDS byte addresses for this thread's staging slots (per buffer)
    const unsigned slot   = (unsigned)(ldr * LDS_STRIDE + seg) * 2u;  // bytes
    const unsigned a_lds0 = (unsigned)(size_t)&As[0][0] + slot;
    const unsigned a_lds1 = (unsigned)(size_t)&As[1][0] + slot;
    const unsigned w_lds0 = (unsigned)(size_t)&Bs[0][0] + slot;
    const unsigned w_lds1 = (unsigned)(size_t)&Bs[1][0] + slot;
    const unsigned dump   = (unsigned)(size_t)&Dump[0] + (unsigned)tid * 32u;

    // zero-fill slots of out-of-range rows ONCE (async never writes them)
    if (!a_ok) {
        v8h z = {};
        *(v8h*)&As[0][ldr * LDS_STRIDE + seg]     = z;
        *(v8h*)&As[0][ldr * LDS_STRIDE + seg + 8] = z;
        *(v8h*)&As[1][ldr * LDS_STRIDE + seg]     = z;
        *(v8h*)&As[1][ldr * LDS_STRIDE + seg + 8] = z;
    }
    if (!w_ok) {
        v8h z = {};
        *(v8h*)&Bs[0][ldr * LDS_STRIDE + seg]     = z;
        *(v8h*)&Bs[0][ldr * LDS_STRIDE + seg + 8] = z;
        *(v8h*)&Bs[1][ldr * LDS_STRIDE + seg]     = z;
        *(v8h*)&Bs[1][ldr * LDS_STRIDE + seg + 8] = z;
    }

    v8f acc[4] = {};

    const int nsteps = K >> 5;

    // EXEC-uniform stage: every thread always issues exactly 4 async copies,
    // so per-wave ASYNCcnt advances identically (wait<=4 == "current done").
    auto stage = [&](int k0, int buf) {
        unsigned al = buf ? a_lds1 : a_lds0;
        unsigned wl = buf ? w_lds1 : w_lds0;
        async_cp16(a_ok ? al      : dump,      aptr + k0);
        async_cp16(a_ok ? al + 16 : dump + 16, aptr + k0 + 8);
        async_cp16(w_ok ? wl      : dump,      wptr + k0);
        async_cp16(w_ok ? wl + 16 : dump + 16, wptr + k0 + 8);
    };

    stage(0, 0);

    for (int i = 0; i < nsteps; ++i) {
        const int cur = i & 1;
        const bool hasnext = (i + 1) < nsteps;
        if (hasnext) {
            stage((i + 1) << 5, cur ^ 1);
            asm volatile("s_wait_asynccnt 0x4" ::: "memory");  // current stage landed
        } else {
            asm volatile("s_wait_asynccnt 0x0" ::: "memory");
        }
        __syncthreads();   // all waves' tiles (and prologue zeros) visible

        // Load ALL fragments first (A + 4x B) so the 4 WMMAs can pipeline
        // behind a single dscnt wait instead of stalling per-MMA.
        // ISA 16-bit A/B layout: lane-half h holds K in {8h..8h+7, 16+8h..23+8h}.
        const _Float16* ap = &As[cur][(wave * 16 + lr) * LDS_STRIDE];
        v8h a_lo = *(const v8h*)(ap + 8 * half);
        v8h a_hi = *(const v8h*)(ap + 16 + 8 * half);
        v16h afrag = __builtin_shufflevector(a_lo, a_hi,
                         0,1,2,3,4,5,6,7,8,9,10,11,12,13,14,15);

        v16h bfrag[4];
#pragma unroll
        for (int nt = 0; nt < 4; ++nt) {
            const _Float16* bp = &Bs[cur][(nt * 16 + lr) * LDS_STRIDE];
            v8h b_lo = *(const v8h*)(bp + 8 * half);
            v8h b_hi = *(const v8h*)(bp + 16 + 8 * half);
            bfrag[nt] = __builtin_shufflevector(b_lo, b_hi,
                            0,1,2,3,4,5,6,7,8,9,10,11,12,13,14,15);
        }

#pragma unroll
        for (int nt = 0; nt < 4; ++nt) {
            acc[nt] = __builtin_amdgcn_wmma_f32_16x16x32_f16(
                false, afrag, false, bfrag[nt], (short)0, acc[nt], false, false);
        }
        __syncthreads();   // done reading buf `cur` before it is refilled
    }

    // C/D f32 layout: VGPR i -> row (i + 8*half), col = lane%16
#pragma unroll
    for (int nt = 0; nt < 4; ++nt) {
        int col = n0 + nt * 16 + lr;
        if (col >= N) continue;
        float bv = bias ? bias[col] : 0.0f;
#pragma unroll
        for (int i = 0; i < 8; ++i) {
            int row = m0 + wave * 16 + i + half * 8;
            if (row >= M) continue;
            float v = acc[nt][i] + bv;
            if (relu) v = fmaxf(v, 0.0f);
            if (o32) o32[(size_t)row * ldo32 + col] = v;
            if (o16) o16[(size_t)row * ldo16 + col] = (_Float16)v;
        }
    }
}

// ---------------------------------------------------------------------------
// f32 -> f16 with zero K-padding: out[rows, kp]
// ---------------------------------------------------------------------------
__global__ void cvt_pad_kernel(const float* __restrict__ in, _Float16* __restrict__ out,
                               int rows, int k, int kp)
{
    size_t idx = (size_t)blockIdx.x * blockDim.x + threadIdx.x;
    size_t tot = (size_t)rows * kp;
    if (idx >= tot) return;
    int r = (int)(idx / kp), c = (int)(idx % kp);
    out[idx] = (c < k) ? (_Float16)in[(size_t)r * k + c] : (_Float16)0.0f;
}

// ---------------------------------------------------------------------------
// BatchNorm1d training-mode stats over all rows (biased var), then apply+ReLU
// ---------------------------------------------------------------------------
__global__ void bn_stats_kernel(const float* __restrict__ x, int rows, int C,
                                float* __restrict__ mu, float* __restrict__ rstd)
{
    int c = blockIdx.x, tid = threadIdx.x;
    __shared__ float ss[256], sq[256];
    float s = 0.f, q = 0.f;
    for (int r = tid; r < rows; r += 256) {
        float v = x[(size_t)r * C + c];
        s += v; q += v * v;
    }
    ss[tid] = s; sq[tid] = q;
    __syncthreads();
    for (int st = 128; st > 0; st >>= 1) {
        if (tid < st) { ss[tid] += ss[tid + st]; sq[tid] += sq[tid + st]; }
        __syncthreads();
    }
    if (tid == 0) {
        float m = ss[0] / (float)rows;
        float var = sq[0] / (float)rows - m * m;
        mu[c] = m;
        rstd[c] = rsqrtf(fmaxf(var, 0.f) + 1e-5f);
    }
}

__global__ void bn_apply_kernel(const float* __restrict__ x, float* __restrict__ y,
                                const float* __restrict__ mu, const float* __restrict__ rstd,
                                const float* __restrict__ g, const float* __restrict__ bt,
                                int rows, int C)
{
    size_t idx = (size_t)blockIdx.x * blockDim.x + threadIdx.x;
    size_t tot = (size_t)rows * C;
    if (idx >= tot) return;
    int c = (int)(idx % C);
    float v = (x[idx] - mu[c]) * rstd[c] * g[c] + bt[c];
    y[idx] = fmaxf(v, 0.f);
}

// ---------------------------------------------------------------------------
// Brute-force kNN: one thread per query, register top-17 insertion sort.
// ---------------------------------------------------------------------------
__global__ void knn_kernel(const float* __restrict__ q, const float* __restrict__ ref,
                           int* __restrict__ out, int B, int Mq, int Nref, int k)
{
    size_t t = (size_t)blockIdx.x * blockDim.x + threadIdx.x;
    if (t >= (size_t)B * Mq) return;
    int b = (int)(t / Mq);
    const float* qp = q + t * 3;
    float qx = qp[0], qy = qp[1], qz = qp[2];
    const float* rp = ref + (size_t)b * Nref * 3;

    float dd[17]; int ii[17];
#pragma unroll
    for (int u = 0; u < 17; ++u) { dd[u] = 3.4e38f; ii[u] = 0; }

    for (int j = 0; j < Nref; ++j) {
        float dx = qx - rp[j * 3 + 0];
        float dy = qy - rp[j * 3 + 1];
        float dz = qz - rp[j * 3 + 2];
        float d = dx * dx + dy * dy + dz * dz;
        if (d < dd[16]) {
            dd[16] = d; ii[16] = j;
#pragma unroll
            for (int u = 16; u > 0; --u) {
                if (dd[u] < dd[u - 1]) {
                    float tf = dd[u]; dd[u] = dd[u - 1]; dd[u - 1] = tf;
                    int   ti = ii[u]; ii[u] = ii[u - 1]; ii[u - 1] = ti;
                }
            }
        }
    }
    int* op = out + t * (size_t)k;
#pragma unroll
    for (int u = 0; u < 17; ++u) if (u < k) op[u] = ii[u];
}

// ---------------------------------------------------------------------------
// Farthest point sampling, one block per batch, deterministic start index 0.
// ---------------------------------------------------------------------------
__global__ void fps_kernel(const float* __restrict__ pos, int N, int m,
                           int* __restrict__ idx, float* __restrict__ mind)
{
    int b = blockIdx.x, tid = threadIdx.x;
    const float* p = pos + (size_t)b * N * 3;
    float* md = mind + (size_t)b * N;
    __shared__ float sv[256];
    __shared__ int   si[256];
    __shared__ float cpt[3];

    float x0 = p[0], y0 = p[1], z0 = p[2];
    for (int i = tid; i < N; i += 256) {
        float dx = p[i * 3] - x0, dy = p[i * 3 + 1] - y0, dz = p[i * 3 + 2] - z0;
        md[i] = dx * dx + dy * dy + dz * dz;
    }
    if (tid == 0) idx[(size_t)b * m] = 0;
    __syncthreads();

    for (int s = 1; s < m; ++s) {
        float bv = -1.f; int bi = 0x7fffffff;
        for (int i = tid; i < N; i += 256) {
            float v = md[i];
            if (v > bv || (v == bv && i < bi)) { bv = v; bi = i; }
        }
        sv[tid] = bv; si[tid] = bi;
        __syncthreads();
        for (int st = 128; st > 0; st >>= 1) {
            if (tid < st) {
                if (sv[tid + st] > sv[tid] ||
                    (sv[tid + st] == sv[tid] && si[tid + st] < si[tid])) {
                    sv[tid] = sv[tid + st]; si[tid] = si[tid + st];
                }
            }
            __syncthreads();
        }
        if (tid == 0) {
            int nx = si[0];
            idx[(size_t)b * m + s] = nx;
            cpt[0] = p[nx * 3]; cpt[1] = p[nx * 3 + 1]; cpt[2] = p[nx * 3 + 2];
        }
        __syncthreads();
        float cx = cpt[0], cy = cpt[1], cz = cpt[2];
        for (int i = tid; i < N; i += 256) {
            float dx = p[i * 3] - cx, dy = p[i * 3 + 1] - cy, dz = p[i * 3 + 2] - cz;
            float d = dx * dx + dy * dy + dz * dz;
            if (d < md[i]) md[i] = d;
        }
        __syncthreads();
    }
}

__global__ void gather_pos_kernel(const float* __restrict__ pos, const int* __restrict__ idx,
                                  float* __restrict__ out, int B, int N, int m)
{
    size_t t = (size_t)blockIdx.x * blockDim.x + threadIdx.x;
    if (t >= (size_t)B * m) return;
    int b = (int)(t / m);
    int j = idx[t];
    const float* src = pos + ((size_t)b * N + j) * 3;
    out[t * 3 + 0] = src[0];
    out[t * 3 + 1] = src[1];
    out[t * 3 + 2] = src[2];
}

// TransitionDown scatter-max equivalent: out[b,i,c] = max_k h[b, nbr[b,i,k], c]
__global__ void gather_max_kernel(const float* __restrict__ h, const int* __restrict__ nbr,
                                  float* __restrict__ out, int B, int N, int m, int C)
{
    size_t idx = (size_t)blockIdx.x * blockDim.x + threadIdx.x;
    size_t tot = (size_t)B * m * C;
    if (idx >= tot) return;
    int c = (int)(idx % C);
    size_t t = idx / C;
    int b = (int)(t / m);
    const int* nb = nbr + t * 16;
    float best = -3.4e38f;
#pragma unroll
    for (int k = 0; k < 16; ++k) {
        float v = h[((size_t)b * N + nb[k]) * C + c];
        best = fmaxf(best, v);
    }
    out[idx] = best;
}

// pos_nn input rows: posdiff[b,n,k,:3] zero-padded to 32 f16 columns
__global__ void build_posdiff_kernel(const float* __restrict__ pos, const int* __restrict__ nbr,
                                     _Float16* __restrict__ out, int B, int Nl)
{
    size_t row = (size_t)blockIdx.x * blockDim.x + threadIdx.x;
    size_t tot = (size_t)B * Nl * 17;
    if (row >= tot) return;
    size_t pt = row / 17;
    int b = (int)(pt / Nl);
    int j = nbr[row];
    const float* pn = pos + pt * 3;
    const float* pj = pos + ((size_t)b * Nl + j) * 3;
    _Float16* o = out + row * 32;
    o[0] = (_Float16)(pn[0] - pj[0]);
    o[1] = (_Float16)(pn[1] - pj[1]);
    o[2] = (_Float16)(pn[2] - pj[2]);
#pragma unroll
    for (int c = 3; c < 32; ++c) o[c] = (_Float16)0.0f;
}

// attn_nn input rows: q[pt] - s[nbr] + delta[row]  (stored f16, stride C)
__global__ void build_alphain_kernel(const float* __restrict__ q, const float* __restrict__ s,
                                     const float* __restrict__ delta, const int* __restrict__ nbr,
                                     _Float16* __restrict__ out, int B, int Nl, int C)
{
    size_t idx = (size_t)blockIdx.x * blockDim.x + threadIdx.x;
    size_t tot = (size_t)B * Nl * 17 * (size_t)C;
    if (idx >= tot) return;
    size_t row = idx / C;
    int c = (int)(idx % C);
    size_t pt = row / 17;
    int b = (int)(pt / Nl);
    int j = nbr[row];
    float v = q[pt * C + c] - s[((size_t)b * Nl + j) * C + c] + delta[idx];
    out[idx] = (_Float16)v;
}

// softmax over the 17 neighbors (per b,n,c) + weighted sum of (v_nbr + delta)
__global__ void softmax_agg_kernel(const float* __restrict__ alpha, const float* __restrict__ v,
                                   const float* __restrict__ delta, const int* __restrict__ nbr,
                                   float* __restrict__ y, int B, int Nl, int C)
{
    size_t idx = (size_t)blockIdx.x * blockDim.x + threadIdx.x;
    size_t tot = (size_t)B * Nl * (size_t)C;
    if (idx >= tot) return;
    size_t pt = idx / C;
    int c = (int)(idx % C);
    int b = (int)(pt / Nl);
    size_t rb = pt * 17;

    float a[17];
    float mx = -3.4e38f;
#pragma unroll
    for (int k = 0; k < 17; ++k) {
        a[k] = alpha[(rb + k) * C + c];
        mx = fmaxf(mx, a[k]);
    }
    float se = 0.f;
#pragma unroll
    for (int k = 0; k < 17; ++k) { a[k] = __expf(a[k] - mx); se += a[k]; }
    float inv = 1.f / se;
    float acc = 0.f;
#pragma unroll
    for (int k = 0; k < 17; ++k) {
        int j = nbr[rb + k];
        float vv = v[((size_t)b * Nl + j) * C + c] + delta[(rb + k) * C + c];
        acc += a[k] * vv;
    }
    y[idx] = acc * inv;
}

// knn_interpolate(k=3): inverse-squared-distance weights
__global__ void interp3_kernel(const float* __restrict__ xsub, const float* __restrict__ posf,
                               const float* __restrict__ poss, const int* __restrict__ nbr3,
                               float* __restrict__ out, int B, int Nf, int Nsub, int C)
{
    size_t idx = (size_t)blockIdx.x * blockDim.x + threadIdx.x;
    size_t tot = (size_t)B * Nf * (size_t)C;
    if (idx >= tot) return;
    size_t pt = idx / C;
    int c = (int)(idx % C);
    int b = (int)(pt / Nf);
    const float* pf = posf + pt * 3;
    float num = 0.f, den = 0.f;
#pragma unroll
    for (int k = 0; k < 3; ++k) {
        int j = nbr3[pt * 3 + k];
        const float* ps = poss + ((size_t)b * Nsub + j) * 3;
        float dx = pf[0] - ps[0], dy = pf[1] - ps[1], dz = pf[2] - ps[2];
        float d2 = dx * dx + dy * dy + dz * dz;
        float w = 1.f / fmaxf(d2, 1e-16f);
        num += w * xsub[((size_t)b * Nsub + j) * C + c];
        den += w;
    }
    out[idx] = num / den;
}

__global__ void add_kernel(const float* __restrict__ a, const float* __restrict__ b,
                           float* __restrict__ o, size_t n)
{
    size_t i = (size_t)blockIdx.x * blockDim.x + threadIdx.x;
    if (i < n) o[i] = a[i] + b[i];
}

// ===========================================================================
// Host orchestration
// ===========================================================================
namespace {

constexpr int Bn = 8, N0 = 4096, CINC = 6, KNN_K = 16, KL = 17, OUTC = 13;
constexpr int DIMS_[4] = {64, 128, 256, 512};

struct Arena {
    char* base; size_t off; size_t cap;
    void* alloc(size_t bytes) {
        size_t a = (off + 255) & ~(size_t)255;
        off = a + bytes;
        return base + a;
    }
    float*     f32(size_t n) { return (float*)alloc(n * 4); }
    _Float16*  f16(size_t n) { return (_Float16*)alloc(n * 2); }
    int*       i32(size_t n) { return (int*)alloc(n * 4); }
};

struct LinP { const float* w; const float* b; int co, ci; };
struct BnP  { const float* w; const float* b; const float* g; const float* bt; int co, ci; };
struct TbP  { LinP a0, a1, li, lo, p0, p1; const float *wdst, *wlin, *wsrc; int c; };

struct LinW { _Float16* w16; const float* b; int co, ci, kp; };
struct BnW  { _Float16* w16; const float* b; const float* g; const float* bt; int co, ci, kp; };
struct TbW  { LinW a0, a1, li, lo, p0, p1, wd, wl, wsc; int c; };

} // namespace

extern "C" void kernel_launch(void* const* d_in, const int* in_sizes, int n_in,
                              void* d_out, int out_size, void* d_ws, size_t ws_size,
                              hipStream_t stream)
{
    (void)in_sizes; (void)n_in; (void)out_size;
    Arena ws{(char*)d_ws, 0, ws_size};

    // ---- parse leaves in jax tree order (dict keys sorted recursively):
    // params{head, mlp_input, mlp_summit, t_down, t_in, t_sum, t_up, td, tu}, pos, x
    int li = 0;
    auto nextp = [&]() -> const float* { return (const float*)d_in[li++]; };
    auto getLin = [&](int co, int ci) -> LinP {
        LinP l; l.w = nextp(); l.b = nextp(); l.co = co; l.ci = ci; return l;
    };
    auto getBn = [&](int co, int ci) -> BnP {
        BnP p; p.w = nextp(); p.b = nextp(); p.g = nextp(); p.bt = nextp();
        p.co = co; p.ci = ci; return p;
    };
    auto getTb = [&](int c) -> TbP {
        TbP t; t.c = c;
        t.a0 = getLin(64, c); t.a1 = getLin(c, 64);          // attn_nn
        t.li = getLin(c, c);  t.lo = getLin(c, c);           // lin_in, lin_out
        t.p0 = getLin(64, 3); t.p1 = getLin(c, 64);          // pos_nn
        t.wdst = nextp(); t.wlin = nextp(); t.wsrc = nextp();
        return t;
    };

    LinP head_p[4];
    head_p[0] = getLin(128, 64);  head_p[1] = getLin(128, 128);
    head_p[2] = getLin(128, 128); head_p[3] = getLin(OUTC, 128);
    BnP  mlp_input_p = getBn(64, CINC);
    LinP summit_p    = getLin(512, 512);
    TbP  t_down_p[3]; for (int i = 0; i < 3; ++i) t_down_p[i] = getTb(DIMS_[i + 1]);
    TbP  t_in_p  = getTb(DIMS_[0]);
    TbP  t_sum_p = getTb(DIMS_[3]);
    TbP  t_up_p[3];   for (int j = 0; j < 3; ++j) t_up_p[j] = getTb(DIMS_[j]);
    BnP  td_p[3];     for (int i = 0; i < 3; ++i) td_p[i] = getBn(DIMS_[i + 1], DIMS_[i]);
    BnP  tu_mlp_p[3], tu_sub_p[3];
    for (int j = 0; j < 3; ++j) {
        tu_mlp_p[j] = getBn(DIMS_[j], DIMS_[j]);
        tu_sub_p[j] = getBn(DIMS_[j], DIMS_[j + 1]);
    }
    const float* pos0 = nextp();
    const float* xin  = nextp();

    // ---- helpers ----
    auto cvt16 = [&](const float* src, int rows, int k) -> _Float16* {
        int kp = kpad32(k);
        _Float16* dst = ws.f16((size_t)rows * kp);
        size_t tot = (size_t)rows * kp;
        cvt_pad_kernel<<<(unsigned)((tot + 255) / 256), 256, 0, stream>>>(src, dst, rows, k, kp);
        return dst;
    };
    auto mkLinW = [&](const LinP& l) -> LinW {
        LinW o; o.co = l.co; o.ci = l.ci; o.kp = kpad32(l.ci);
        o.w16 = cvt16(l.w, l.co, l.ci); o.b = l.b; return o;
    };
    auto mkLinW_nb = [&](const float* w, int co, int ci) -> LinW {
        LinW o; o.co = co; o.ci = ci; o.kp = kpad32(ci);
        o.w16 = cvt16(w, co, ci); o.b = nullptr; return o;
    };
    auto mkBnW = [&](const BnP& p) -> BnW {
        BnW o; o.co = p.co; o.ci = p.ci; o.kp = kpad32(p.ci);
        o.w16 = cvt16(p.w, p.co, p.ci); o.b = p.b; o.g = p.g; o.bt = p.bt; return o;
    };
    auto mkTbW = [&](const TbP& t) -> TbW {
        TbW w; w.c = t.c;
        w.a0 = mkLinW(t.a0); w.a1 = mkLinW(t.a1);
        w.li = mkLinW(t.li); w.lo = mkLinW(t.lo);
        w.p0 = mkLinW(t.p0); w.p1 = mkLinW(t.p1);
        w.wd = mkLinW_nb(t.wdst, t.c, t.c);
        w.wl = mkLinW_nb(t.wlin, t.c, t.c);
        w.wsc = mkLinW_nb(t.wsrc, t.c, t.c);
        return w;
    };
    auto gemm = [&](const _Float16* A, int lda, const LinW& w,
                    float* o32, int ldo32, _Float16* o16, int ldo16, int M, int relu) {
        dim3 g((unsigned)((w.co + 63) / 64), (unsigned)((M + 63) / 64));
        wmma_gemm_kernel<<<g, 128, 0, stream>>>(A, lda, w.w16, w.kp, w.b,
                                                o32, ldo32, o16, ldo16, M, w.co, w.kp, relu);
    };
    auto mlp_bn_run = [&](const float* x, int M, const BnW& p, float* out) {
        size_t mark = ws.off;
        _Float16* x16 = cvt16(x, M, p.ci);
        float* lin = ws.f32((size_t)M * p.co);
        LinW lw{p.w16, p.b, p.co, p.ci, p.kp};
        gemm(x16, p.kp, lw, lin, p.co, nullptr, 0, M, 0);
        float* mu = ws.f32(p.co);
        float* rs = ws.f32(p.co);
        bn_stats_kernel<<<p.co, 256, 0, stream>>>(lin, M, p.co, mu, rs);
        size_t tot = (size_t)M * p.co;
        bn_apply_kernel<<<(unsigned)((tot + 255) / 256), 256, 0, stream>>>(
            lin, out, mu, rs, p.g, p.bt, M, p.co);
        ws.off = mark;
    };
    auto knn_run = [&](const float* qp, int Mq, const float* rp, int Nr, int k) -> int* {
        int* o = ws.i32((size_t)Bn * Mq * k);
        size_t tot = (size_t)Bn * Mq;
        knn_kernel<<<(unsigned)((tot + 255) / 256), 256, 0, stream>>>(qp, rp, o, Bn, Mq, Nr, k);
        return o;
    };
    auto tblock_run = [&](const float* x, const float* pos, const int* nbr,
                          int Nl, const TbW& p, float* out) {
        int M = Bn * Nl;
        size_t R = (size_t)M * KL;
        int C = p.c;
        size_t mark = ws.off;

        _Float16* x16 = cvt16(x, M, C);
        _Float16* x1 = ws.f16((size_t)M * C);
        gemm(x16, C, p.li, nullptr, 0, x1, C, M, 1);             // lin_in + ReLU

        float* q = ws.f32((size_t)M * C); gemm(x1, C, p.wd,  q, C, nullptr, 0, M, 0);
        float* s = ws.f32((size_t)M * C); gemm(x1, C, p.wsc, s, C, nullptr, 0, M, 0);
        float* v = ws.f32((size_t)M * C); gemm(x1, C, p.wl,  v, C, nullptr, 0, M, 0);

        _Float16* pd = ws.f16(R * 32);
        build_posdiff_kernel<<<(unsigned)((R + 255) / 256), 256, 0, stream>>>(pos, nbr, pd, Bn, Nl);
        _Float16* ph = ws.f16(R * 64);
        gemm(pd, 32, p.p0, nullptr, 0, ph, 64, (int)R, 1);       // pos_nn L1 + ReLU
        float* delta = ws.f32(R * (size_t)C);
        gemm(ph, 64, p.p1, delta, C, nullptr, 0, (int)R, 1);     // pos_nn L2 + ReLU

        _Float16* ai = ws.f16(R * (size_t)C);
        {
            size_t tot = R * (size_t)C;
            build_alphain_kernel<<<(unsigned)((tot + 255) / 256), 256, 0, stream>>>(
                q, s, delta, nbr, ai, Bn, Nl, C);
        }
        _Float16* ah = ws.f16(R * 64);
        gemm(ai, C, p.a0, nullptr, 0, ah, 64, (int)R, 1);        // attn_nn L1 + ReLU
        float* alpha = ws.f32(R * (size_t)C);
        gemm(ah, 64, p.a1, alpha, C, nullptr, 0, (int)R, 1);     // attn_nn L2 + ReLU

        float* y = ws.f32((size_t)M * C);
        {
            size_t tot = (size_t)M * C;
            softmax_agg_kernel<<<(unsigned)((tot + 255) / 256), 256, 0, stream>>>(
                alpha, v, delta, nbr, y, Bn, Nl, C);
        }
        _Float16* y16 = cvt16(y, M, C);
        gemm(y16, C, p.lo, out, C, nullptr, 0, M, 1);            // lin_out + ReLU
        ws.off = mark;
    };

    // ---- one-time weight conversion (persistent in arena) ----
    BnW  mlp_input_w = mkBnW(mlp_input_p);
    LinW summit_w    = mkLinW(summit_p);
    TbW  t_in_w  = mkTbW(t_in_p);
    TbW  t_sum_w = mkTbW(t_sum_p);
    TbW  t_down_w[3]; for (int i = 0; i < 3; ++i) t_down_w[i] = mkTbW(t_down_p[i]);
    TbW  t_up_w[3];   for (int j = 0; j < 3; ++j) t_up_w[j] = mkTbW(t_up_p[j]);
    BnW  td_w[3];     for (int i = 0; i < 3; ++i) td_w[i] = mkBnW(td_p[i]);
    BnW  tu_mlp_w[3], tu_sub_w[3];
    for (int j = 0; j < 3; ++j) { tu_mlp_w[j] = mkBnW(tu_mlp_p[j]); tu_sub_w[j] = mkBnW(tu_sub_p[j]); }
    LinW head_w[4]; for (int h = 0; h < 4; ++h) head_w[h] = mkLinW(head_p[h]);

    // ---- forward ----
    const float* poss[4]; int Ns_[4] = {N0, N0 / 4, N0 / 16, N0 / 64};
    int*   knn17[4];
    float* xs[4];
    poss[0] = pos0;

    knn17[0] = knn_run(poss[0], Ns_[0], poss[0], Ns_[0], KL);
    float* feat0 = ws.f32((size_t)Bn * N0 * 64);
    mlp_bn_run(xin, Bn * N0, mlp_input_w, feat0);
    xs[0] = ws.f32((size_t)Bn * N0 * 64);
    tblock_run(feat0, poss[0], knn17[0], N0, t_in_w, xs[0]);

    float* cur = xs[0];
    for (int i = 0; i < 3; ++i) {
        int Nl = Ns_[i], m = Ns_[i + 1], Cn = DIMS_[i + 1];
        int*   fidx = ws.i32((size_t)Bn * m);
        float* mind = ws.f32((size_t)Bn * Nl);
        fps_kernel<<<Bn, 256, 0, stream>>>(poss[i], Nl, m, fidx, mind);
        float* spos = ws.f32((size_t)Bn * m * 3);
        gather_pos_kernel<<<(unsigned)(((size_t)Bn * m + 255) / 256), 256, 0, stream>>>(
            poss[i], fidx, spos, Bn, Nl, m);
        poss[i + 1] = spos;
        int* nbrd = knn_run(spos, m, poss[i], Nl, KNN_K);
        float* h = ws.f32((size_t)Bn * Nl * Cn);
        mlp_bn_run(cur, Bn * Nl, td_w[i], h);
        float* xg = ws.f32((size_t)Bn * m * Cn);
        {
            size_t tot = (size_t)Bn * m * Cn;
            gather_max_kernel<<<(unsigned)((tot + 255) / 256), 256, 0, stream>>>(
                h, nbrd, xg, Bn, Nl, m, Cn);
        }
        knn17[i + 1] = knn_run(spos, m, spos, m, KL);
        xs[i + 1] = ws.f32((size_t)Bn * m * Cn);
        tblock_run(xg, spos, knn17[i + 1], m, t_down_w[i], xs[i + 1]);
        cur = xs[i + 1];
    }

    // summit: mlp_nobn (ReLU) + t_sum
    {
        int M3 = Bn * Ns_[3];
        _Float16* x16 = cvt16(cur, M3, 512);
        float* t = ws.f32((size_t)M3 * 512);
        gemm(x16, 512, summit_w, t, 512, nullptr, 0, M3, 1);
        float* o = ws.f32((size_t)M3 * 512);
        tblock_run(t, poss[3], knn17[3], Ns_[3], t_sum_w, o);
        cur = o;
    }

    // up path
    for (int i = 0; i < 3; ++i) {
        int j = 2 - i;
        int Nf = Ns_[j], Nsub = Ns_[j + 1], Cf = DIMS_[j];
        float* xsub = ws.f32((size_t)Bn * Nsub * Cf);
        mlp_bn_run(cur, Bn * Nsub, tu_sub_w[j], xsub);
        int* nbr3 = knn_run(poss[j], Nf, poss[j + 1], Nsub, 3);
        float* xint = ws.f32((size_t)Bn * Nf * Cf);
        {
            size_t tot = (size_t)Bn * Nf * Cf;
            interp3_kernel<<<(unsigned)((tot + 255) / 256), 256, 0, stream>>>(
                xsub, poss[j], poss[j + 1], nbr3, xint, Bn, Nf, Nsub, Cf);
        }
        float* xf = ws.f32((size_t)Bn * Nf * Cf);
        mlp_bn_run(xs[j], Bn * Nf, tu_mlp_w[j], xf);
        {
            size_t tot = (size_t)Bn * Nf * Cf;
            add_kernel<<<(unsigned)((tot + 255) / 256), 256, 0, stream>>>(xf, xint, xf, tot);
        }
        float* o = ws.f32((size_t)Bn * Nf * Cf);
        tblock_run(xf, poss[j], knn17[j], Nf, t_up_w[j], o);
        cur = o;
    }

    // head: 3x ReLU GEMM (f16 chained), final GEMM -> d_out f32 [B*N, 13]
    {
        int M = Bn * N0;
        _Float16* h0 = cvt16(cur, M, 64);
        _Float16* h1 = ws.f16((size_t)M * 128);
        gemm(h0, 64, head_w[0], nullptr, 0, h1, 128, M, 1);
        _Float16* h2 = ws.f16((size_t)M * 128);
        gemm(h1, 128, head_w[1], nullptr, 0, h2, 128, M, 1);
        _Float16* h3 = ws.f16((size_t)M * 128);
        gemm(h2, 128, head_w[2], nullptr, 0, h3, 128, M, 1);
        gemm(h3, 128, head_w[3], (float*)d_out, OUTC, nullptr, 0, M, 0);
    }
}
